// MSARowAttentionWithPairBias_88819923681828
// MI455X (gfx1250) — matrix operand order repaired and checked by reading, not compiled, not run on hardware
//
#include <hip/hip_runtime.h>
#include <hip/hip_bf16.h>

// S=64, R=384, CM=256, CP=128, H=8, C=32
#define S_DIM 64
#define R_DIM 384
#define CM_DIM 256
#define CP_DIM 128
#define H_DIM 8
#define C_DIM 32
#define SR (S_DIM * R_DIM)            // 24576 rows
#define PAIRS (R_DIM * R_DIM)         // 147456

typedef __attribute__((ext_vector_type(16))) _Float16 v16h;
typedef __attribute__((ext_vector_type(8)))  _Float16 v8h;
typedef __attribute__((ext_vector_type(8)))  float    v8f;
typedef __attribute__((ext_vector_type(4)))  int      v4i;

// ---------------------------------------------------------------------------
// gfx1250 async global->LDS staging (ASYNCcnt path). Guarded so a missing
// builtin falls back to plain load/store staging.
// Prototype (from hipcc diagnostic): (int4 AS1* gsrc, int4 AS3* ldst, Ii, Ii)
// ---------------------------------------------------------------------------
#if defined(__has_builtin)
#  if __has_builtin(__builtin_amdgcn_global_load_async_to_lds_b128) && \
      __has_builtin(__builtin_amdgcn_s_wait_asynccnt)
#    define HAVE_ASYNC_LDS 1
#  else
#    define HAVE_ASYNC_LDS 0
#  endif
#else
#  define HAVE_ASYNC_LDS 0
#endif

typedef __attribute__((address_space(1))) v4i as1_v4i;
typedef __attribute__((address_space(3))) v4i as3_v4i;

__device__ inline void stage16(_Float16* ldst, const _Float16* gsrc) {
#if HAVE_ASYNC_LDS
  // generic->LDS pointer: low 32 bits of a generic LDS address are the LDS
  // byte offset on amdgcn; generic global pointers are numerically identical
  // to AS1 pointers. Route through integers (always-legal casts).
  as1_v4i* g = (as1_v4i*)(unsigned long long)gsrc;
  as3_v4i* l = (as3_v4i*)(unsigned int)(unsigned long long)ldst;
  __builtin_amdgcn_global_load_async_to_lds_b128(g, l, 0, 0);
#else
  *(v8h*)ldst = *(const v8h*)gsrc;
#endif
}

__device__ inline void stage_wait() {
#if HAVE_ASYNC_LDS
  __builtin_amdgcn_s_wait_asynccnt(0);
#endif
  __syncthreads();
}

// ---------------------------------------------------------------------------
// A-fragment loader for v_wmma_f32_16x16x32_f16 from a row-major f16 matrix.
//   lanes 0-15  : M=lane,    halves[0..15]=K 0..15
//   lanes 16-31 : M=lane-16, halves[0..15]=K 16..31
// ---------------------------------------------------------------------------
__device__ inline v16h load_a_rowmajor(const _Float16* __restrict__ p, int ld) {
  const int lane = threadIdx.x & 31;
  const int hg   = lane >> 4;
  const int m    = lane & 15;
  const _Float16* row = p + (size_t)m * ld + hg * 16;
  const v8h lo = *(const v8h*)(row);
  const v8h hi = *(const v8h*)(row + 8);
  v16h a;
#pragma unroll
  for (int i = 0; i < 8; ++i) { a[i] = lo[i]; a[8 + i] = hi[i]; }
  return a;
}

__device__ inline v8f wmma_f16(v16h a, v16h b, v8f c) {
  return __builtin_amdgcn_wmma_f32_16x16x32_f16(false, a, false, b, (short)0, c,
                                                false, false);
}

// ---------------------------------------------------------------------------
// Pack a 256x256 f32 weight (K x N, row-major) into f16 B fragments:
//   lane L: N = nt*16 + (L&15), K = kc*32 + (L>>4)*16 + i
//   dst index = ((kc*16 + nt)*32 + lane)*16 + i
// ---------------------------------------------------------------------------
__global__ __launch_bounds__(256) void pack_b_kernel(const float* __restrict__ W,
                                                     _Float16* __restrict__ dst) {
  const int t    = blockIdx.x * 256 + threadIdx.x;   // 0 .. 65535
  const int i    = t & 15;
  const int lane = (t >> 4) & 31;
  const int nt   = (t >> 9) & 15;
  const int kc   = t >> 13;
  const int N    = nt * 16 + (lane & 15);
  const int K    = kc * 32 + ((lane >> 4) << 4) + i;
  dst[t] = (_Float16)W[(size_t)K * 256 + N];
}

// ---------------------------------------------------------------------------
// LayerNorm over CM=256: one wave per row, 8 floats per lane, f16 output.
// ---------------------------------------------------------------------------
__global__ __launch_bounds__(256) void ln_msa_kernel(const float* __restrict__ msa,
                                                     const float* __restrict__ g,
                                                     const float* __restrict__ b,
                                                     _Float16* __restrict__ xh) {
  const int lane = threadIdx.x & 31;
  const int row  = blockIdx.x * 8 + (threadIdx.x >> 5);
  const float* p = msa + (size_t)row * CM_DIM + lane * 8;
  float x[8];
  const float4 a0 = *(const float4*)(p);
  const float4 a1 = *(const float4*)(p + 4);
  x[0]=a0.x; x[1]=a0.y; x[2]=a0.z; x[3]=a0.w;
  x[4]=a1.x; x[5]=a1.y; x[6]=a1.z; x[7]=a1.w;
  float sum = 0.f, sq = 0.f;
#pragma unroll
  for (int i = 0; i < 8; ++i) { sum += x[i]; sq += x[i] * x[i]; }
#pragma unroll
  for (int m = 1; m <= 16; m <<= 1) {
    sum += __shfl_xor(sum, m, 32);
    sq  += __shfl_xor(sq,  m, 32);
  }
  const float mu  = sum * (1.0f / 256.0f);
  const float var = sq * (1.0f / 256.0f) - mu * mu;
  const float rs  = rsqrtf(var + 1e-5f);
  _Float16* o = xh + (size_t)row * CM_DIM + lane * 8;
#pragma unroll
  for (int i = 0; i < 8; ++i) {
    const int cidx = lane * 8 + i;
    o[i] = (_Float16)((x[i] - mu) * rs * g[cidx] + b[cidx]);
  }
}

// ---------------------------------------------------------------------------
// Pair bias: pb[h][i][j] = LN(pair[i,j,:]) @ Wb[:,h] + bb[h].  Wave per (i,j).
// ---------------------------------------------------------------------------
__global__ __launch_bounds__(256) void pair_bias_kernel(const float* __restrict__ pair,
                                                        const float* __restrict__ pg,
                                                        const float* __restrict__ pbeta,
                                                        const float* __restrict__ Wb,
                                                        const float* __restrict__ bb,
                                                        float* __restrict__ pb) {
  const int lane = threadIdx.x & 31;
  const int wid  = blockIdx.x * 8 + (threadIdx.x >> 5);   // 0 .. 147455
  const float* p = pair + (size_t)wid * CP_DIM + lane * 4;
  const float4 x4 = *(const float4*)p;
  float x[4] = {x4.x, x4.y, x4.z, x4.w};
  float sum = 0.f, sq = 0.f;
#pragma unroll
  for (int i = 0; i < 4; ++i) { sum += x[i]; sq += x[i] * x[i]; }
#pragma unroll
  for (int m = 1; m <= 16; m <<= 1) {
    sum += __shfl_xor(sum, m, 32);
    sq  += __shfl_xor(sq,  m, 32);
  }
  const float mu  = sum * (1.0f / 128.0f);
  const float var = sq * (1.0f / 128.0f) - mu * mu;
  const float rs  = rsqrtf(var + 1e-5f);
  float acc[H_DIM];
#pragma unroll
  for (int h = 0; h < H_DIM; ++h) acc[h] = 0.f;
#pragma unroll
  for (int t = 0; t < 4; ++t) {
    const int cidx = lane * 4 + t;
    const float xn = (x[t] - mu) * rs * pg[cidx] + pbeta[cidx];
#pragma unroll
    for (int h = 0; h < H_DIM; ++h) acc[h] += xn * Wb[(size_t)cidx * H_DIM + h];
  }
#pragma unroll
  for (int h = 0; h < H_DIM; ++h) {
#pragma unroll
    for (int m = 1; m <= 16; m <<= 1) acc[h] += __shfl_xor(acc[h], m, 32);
  }
  if (lane < H_DIM) {
    float v = acc[0];
#pragma unroll
    for (int h = 1; h < H_DIM; ++h) if (lane == h) v = acc[h];
    pb[(size_t)lane * PAIRS + wid] = v + bb[lane];
  }
}

// ---------------------------------------------------------------------------
// Fused QKVG GEMM: x_hat(24576x256) @ [Wq|Wk|Wv|Wg](256x1024), WMMA f16.
// One wave per 16x64 output strip. Per K-step: 1 A-load + 4 B-loads issued
// up front, then 4 WMMAs (partial loadcnt waits, deep pipelining).
// ---------------------------------------------------------------------------
__global__ __launch_bounds__(256) void qkvg_gemm_kernel(const _Float16* __restrict__ xh,
                                                        const _Float16* __restrict__ wpack,
                                                        const float* __restrict__ bg,
                                                        _Float16* __restrict__ qbuf,
                                                        _Float16* __restrict__ kbuf,
                                                        _Float16* __restrict__ vpack,
                                                        _Float16* __restrict__ gbuf) {
  const int lane = threadIdx.x & 31;
  const int wid  = blockIdx.x * 8 + (threadIdx.x >> 5);
  const int mt   = wid >> 4;        // 0 .. 1535
  const int ng   = wid & 15;        // 16 groups of 4 tiles over N=1024
  const int w    = ng >> 2;         // weight index 0..3
  const int ntw0 = (ng & 3) * 4;    // first N-tile within this weight
  const _Float16* abase = xh + (size_t)mt * 16 * CM_DIM;
  const _Float16* bbase = wpack + (size_t)w * 65536;
  v8f acc[4] = {};
#pragma unroll
  for (int kc = 0; kc < 8; ++kc) {
    const v16h a = load_a_rowmajor(abase + kc * 32, CM_DIM);
    v16h b[4];
#pragma unroll
    for (int j = 0; j < 4; ++j)
      b[j] = *(const v16h*)(bbase + ((size_t)(kc * 16 + ntw0 + j) * 32 + lane) * 16);
#pragma unroll
    for (int j = 0; j < 4; ++j) acc[j] = wmma_f16(a, b[j], acc[j]);
  }
  const int hg   = lane >> 4;
  const int ln16 = lane & 15;
#pragma unroll
  for (int r = 0; r < 8; ++r) {
    const int m  = mt * 16 + r + hg * 8;
    const int s  = m / R_DIM;
    const int rr = m - s * R_DIM;
#pragma unroll
    for (int j = 0; j < 4; ++j) {
      const int col = (ntw0 + j) * 16 + ln16;   // column within this weight
      const int h   = col >> 5;
      const int c   = col & 31;
      const float val = acc[j][r];
      if (w == 0) {
        qbuf[(((size_t)s * H_DIM + h) * R_DIM + rr) * C_DIM + c] =
            (_Float16)(val * 0.17677669529663689f);   // 1/sqrt(32)
      } else if (w == 1) {
        kbuf[(((size_t)s * H_DIM + h) * R_DIM + rr) * C_DIM + c] = (_Float16)val;
      } else if (w == 2) {
        const int kchunk = rr >> 5;
        const int kk     = rr & 31;
        const int ctile  = c >> 4;
        const int lp     = (c & 15) + ((kk >> 4) << 4);
        const int ip     = kk & 15;
        vpack[(((((size_t)s * H_DIM + h) * 12 + kchunk) * 2 + ctile) * 32 + lp) * 16 + ip] =
            (_Float16)val;
      } else {
        const float z = val + bg[col];
        gbuf[((size_t)s * R_DIM + rr) * 256 + col] =
            (_Float16)(1.0f / (1.0f + __expf(-z)));
      }
    }
  }
}

// ---------------------------------------------------------------------------
// Flash attention: one block per (s,h). K + packed-V staged into LDS once
// (async global->LDS), 8 waves x 3 q-tiles each. Per 32-key chunk:
// 2 score WMMAs, online softmax, exp(P) staged through per-wave LDS in
// A-fragment layout, 2 PV WMMAs. Gated epilogue.
// ---------------------------------------------------------------------------
__global__ __launch_bounds__(256) void attn_kernel(const _Float16* __restrict__ qbuf,
                                                   const _Float16* __restrict__ kbuf,
                                                   const _Float16* __restrict__ vpack,
                                                   const _Float16* __restrict__ gbuf,
                                                   const float* __restrict__ pb,
                                                   _Float16* __restrict__ att) {
  __shared__ __align__(32) _Float16 Ksh[R_DIM * C_DIM];     // 24 KB
  __shared__ __align__(32) _Float16 Vsh[R_DIM * C_DIM];     // 24 KB (packed)
  __shared__ __align__(32) _Float16 Plds[8][512];           // 8 KB
  const int lane = threadIdx.x & 31;
  const int wv   = threadIdx.x >> 5;
  const int sh   = blockIdx.x;        // == s*H + h
  const int h    = sh & 7;
  const int s    = sh >> 3;
  const _Float16* Qg = qbuf + (size_t)sh * R_DIM * C_DIM;
  const _Float16* Kg = kbuf + (size_t)sh * R_DIM * C_DIM;
  const _Float16* Vg = vpack + (size_t)sh * R_DIM * C_DIM;  // 12288 halves
  const float* pbh   = pb + (size_t)h * PAIRS;
  const int hg   = lane >> 4;
  const int ln16 = lane & 15;
  _Float16* P = Plds[wv];

  __builtin_prefetch(Qg + (size_t)wv * 3 * 16 * C_DIM, 0, 0);

  // ---- stage K and V into LDS: 2 * 1536 16-byte units over 256 threads ----
#pragma unroll
  for (int it = 0; it < 6; ++it) {
    const int u = it * 256 + threadIdx.x;   // 0 .. 1535
    stage16(Ksh + (size_t)u * 8, Kg + (size_t)u * 8);
    stage16(Vsh + (size_t)u * 8, Vg + (size_t)u * 8);
  }
  stage_wait();

  for (int t = 0; t < 3; ++t) {
    const int qt = wv * 3 + t;
    const int q0 = qt * 16;
    const v16h qa = load_a_rowmajor(Qg + (size_t)q0 * C_DIM, C_DIM);
    v8f o0 = {}, o1 = {};
    float mrow[8], lrow[8];
#pragma unroll
    for (int r = 0; r < 8; ++r) { mrow[r] = -1e30f; lrow[r] = 0.f; }

    for (int kc = 0; kc < 12; ++kc) {
      const int k0 = kc * 32;
      // B fragments of K^T from LDS: 16 contiguous halves of one key row
      const v16h kb0 = *(const v16h*)(Ksh + (size_t)(k0 + ln16) * C_DIM + hg * 16);
      const v16h kb1 = *(const v16h*)(Ksh + (size_t)(k0 + 16 + ln16) * C_DIM + hg * 16);
      v8f s0 = {}, s1 = {};
      s0 = wmma_f16(qa, kb0, s0);
      s1 = wmma_f16(qa, kb1, s1);
#pragma unroll
      for (int r = 0; r < 8; ++r) {
        const int row = q0 + r + hg * 8;
        float a0 = s0[r] + pbh[(size_t)row * R_DIM + k0 + ln16];
        float a1 = s1[r] + pbh[(size_t)row * R_DIM + k0 + 16 + ln16];
        float cm = fmaxf(a0, a1);
#pragma unroll
        for (int m = 1; m <= 8; m <<= 1) cm = fmaxf(cm, __shfl_xor(cm, m, 32));
        const float mn    = fmaxf(mrow[r], cm);
        const float alpha = __expf(mrow[r] - mn);
        mrow[r] = mn;
        const float p0 = __expf(a0 - mn);
        const float p1 = __expf(a1 - mn);
        float ps = p0 + p1;
#pragma unroll
        for (int m = 1; m <= 8; m <<= 1) ps += __shfl_xor(ps, m, 32);
        lrow[r] = lrow[r] * alpha + ps;
        o0[r] *= alpha;
        o1[r] *= alpha;
        // stash exp(P) into per-wave LDS at A-fragment position
        const int m16 = r + hg * 8;
        const int l0  = m16;
        P[l0 * 16 + ln16]        = (_Float16)p0;   // K' = ln16
        P[(l0 + 16) * 16 + ln16] = (_Float16)p1;   // K' = 16 + ln16
      }
      const v16h pa = *(const v16h*)(P + lane * 16);
      const _Float16* vb = Vsh + (size_t)kc * 1024;
      const v16h v0 = *(const v16h*)(vb + lane * 16);
      const v16h v1 = *(const v16h*)(vb + 512 + lane * 16);
      o0 = wmma_f16(pa, v0, o0);
      o1 = wmma_f16(pa, v1, o1);
    }
#pragma unroll
    for (int r = 0; r < 8; ++r) {
      const int row   = q0 + r + hg * 8;
      const float inv = 1.0f / lrow[r];
      const size_t base = ((size_t)s * R_DIM + row) * 256 + h * C_DIM;
      const float g0 = (float)gbuf[base + ln16];
      const float g1 = (float)gbuf[base + 16 + ln16];
      att[base + ln16]      = (_Float16)(o0[r] * inv * g0);
      att[base + 16 + ln16] = (_Float16)(o1[r] * inv * g1);
    }
  }
}

// ---------------------------------------------------------------------------
// Output projection: att(24576x256) @ Wp(256x256) + bp -> f32 out.
// One wave per 16x64 strip; loads issued ahead of the 4 WMMAs per K-step.
// ---------------------------------------------------------------------------
__global__ __launch_bounds__(256) void proj_kernel(const _Float16* __restrict__ att,
                                                   const _Float16* __restrict__ wp,
                                                   const float* __restrict__ bp,
                                                   float* __restrict__ out) {
  const int lane = threadIdx.x & 31;
  const int wid  = blockIdx.x * 8 + (threadIdx.x >> 5);
  const int mt   = wid >> 2;        // 0 .. 1535
  const int ng   = wid & 3;         // 4 groups of 4 N-tiles
  const int nt0  = ng * 4;
  const _Float16* abase = att + (size_t)mt * 16 * 256;
  v8f acc[4] = {};
#pragma unroll
  for (int kc = 0; kc < 8; ++kc) {
    const v16h a = load_a_rowmajor(abase + kc * 32, 256);
    v16h b[4];
#pragma unroll
    for (int j = 0; j < 4; ++j)
      b[j] = *(const v16h*)(wp + ((size_t)(kc * 16 + nt0 + j) * 32 + lane) * 16);
#pragma unroll
    for (int j = 0; j < 4; ++j) acc[j] = wmma_f16(a, b[j], acc[j]);
  }
  const int hg   = lane >> 4;
  const int ln16 = lane & 15;
#pragma unroll
  for (int r = 0; r < 8; ++r) {
    const int m = mt * 16 + r + hg * 8;
#pragma unroll
    for (int j = 0; j < 4; ++j) {
      const int n = (nt0 + j) * 16 + ln16;
      out[(size_t)m * 256 + n] = acc[j][r] + bp[n];
    }
  }
}

// ---------------------------------------------------------------------------
// Host-side orchestration
// ---------------------------------------------------------------------------
extern "C" void kernel_launch(void* const* d_in, const int* in_sizes, int n_in,
                              void* d_out, int out_size, void* d_ws, size_t ws_size,
                              hipStream_t stream) {
  (void)in_sizes; (void)n_in; (void)out_size; (void)ws_size;
  const float* msa   = (const float*)d_in[0];
  const float* pair  = (const float*)d_in[1];
  const float* ln_g  = (const float*)d_in[2];
  const float* ln_b  = (const float*)d_in[3];
  const float* Wq    = (const float*)d_in[4];
  const float* Wk    = (const float*)d_in[5];
  const float* Wv    = (const float*)d_in[6];
  const float* Wg    = (const float*)d_in[7];
  const float* bg    = (const float*)d_in[8];
  const float* Wp    = (const float*)d_in[9];
  const float* bp    = (const float*)d_in[10];
  const float* pln_g = (const float*)d_in[11];
  const float* pln_b = (const float*)d_in[12];
  const float* Wb    = (const float*)d_in[13];
  const float* bb    = (const float*)d_in[14];
  float* out = (float*)d_out;

  char* ws = (char*)d_ws;
  _Float16* xh    = (_Float16*)(ws + 0);           // 24576*256 f16
  _Float16* qbuf  = (_Float16*)(ws + 12582912);    // (S,H,R,C) f16
  _Float16* kbuf  = (_Float16*)(ws + 25165824);    // (S,H,R,C) f16
  _Float16* vpack = (_Float16*)(ws + 37748736);    // B-fragment packed f16
  _Float16* gbuf  = (_Float16*)(ws + 50331648);    // (S,R,256) f16
  _Float16* att   = (_Float16*)(ws + 62914560);    // (S,R,256) f16
  _Float16* wqkvg = (_Float16*)(ws + 75497472);    // 4 * 65536 f16
  _Float16* wpp   = (_Float16*)(ws + 76021760);    // 65536 f16
  float*    pbb   = (float*)   (ws + 76152832);    // (H,R,R) f32

  // 1) pack weights into WMMA B-fragment layout
  pack_b_kernel<<<256, 256, 0, stream>>>(Wq, wqkvg + 0 * 65536);
  pack_b_kernel<<<256, 256, 0, stream>>>(Wk, wqkvg + 1 * 65536);
  pack_b_kernel<<<256, 256, 0, stream>>>(Wv, wqkvg + 2 * 65536);
  pack_b_kernel<<<256, 256, 0, stream>>>(Wg, wqkvg + 3 * 65536);
  pack_b_kernel<<<256, 256, 0, stream>>>(Wp, wpp);

  // 2) LayerNorm of MSA rep -> f16
  ln_msa_kernel<<<SR / 8, 256, 0, stream>>>(msa, ln_g, ln_b, xh);

  // 3) pair bias (H,R,R)
  pair_bias_kernel<<<PAIRS / 8, 256, 0, stream>>>(pair, pln_g, pln_b, Wb, bb, pbb);

  // 4) fused QKVG projection GEMM (WMMA, 4 N-tiles per wave)
  qkvg_gemm_kernel<<<(SR / 16) * 16 / 8, 256, 0, stream>>>(xh, wqkvg, bg,
                                                           qbuf, kbuf, vpack, gbuf);

  // 5) flash attention with pair bias + gating (WMMA, async LDS-staged K/V)
  attn_kernel<<<S_DIM * H_DIM, 256, 0, stream>>>(qbuf, kbuf, vpack, gbuf, pbb, att);

  // 6) output projection (WMMA, 4 N-tiles per wave) -> f32
  proj_kernel<<<(SR / 16) * 4 / 8, 256, 0, stream>>>(att, wpp, bp, out);
}
